// AttentionBlock3D_9740985827898
// MI455X (gfx1250) — matrix operand order repaired and checked by reading, not compiled, not run on hardware
//
#include <hip/hip_runtime.h>

typedef __attribute__((ext_vector_type(16))) _Float16 v16h;
typedef __attribute__((ext_vector_type(8)))  _Float16 v8h;
typedef __attribute__((ext_vector_type(2)))  _Float16 h2;
typedef __attribute__((ext_vector_type(8)))  float    v8f;
typedef __attribute__((ext_vector_type(2)))  int      i32x2;

#define C_  64
#define G_  8
#define H_  8
#define D_  8
#define S_  4096
#define EPSF 1e-5f

// ---- CDNA5 async-to-LDS path (probe via __has_builtin; sync fallback) ------
#if defined(__AMDGCN__) && __has_builtin(__builtin_amdgcn_global_load_async_to_lds_b64)
#define HAS_ASYNC_LDS 1
#else
#define HAS_ASYNC_LDS 0
#endif

// b64 variant takes (v2i addrspace(1)* src, v2i addrspace(3)* dst, imm off, imm cpol)
#define GPTR64(p) ((__attribute__((address_space(1))) i32x2*)(p))
#define LPTR64(p) ((__attribute__((address_space(3))) i32x2*)(p))

#if HAS_ASYNC_LDS
#if __has_builtin(__builtin_amdgcn_s_wait_asynccnt)
#define WAIT_ASYNC(n) __builtin_amdgcn_s_wait_asynccnt(n)
#else
#define WAIT_ASYNC(n) asm volatile("s_wait_asynccnt %0" ::"n"(n))
#endif
#else
#define WAIT_ASYNC(n)
#endif

// ---------------------------------------------------------------------------
// Kernel 1: GroupNorm statistics. 8 groups, each = 8 channels * 4096 spatial
// contiguous floats. One block per group.
// ---------------------------------------------------------------------------
__global__ __launch_bounds__(256) void gn_stats(const float* __restrict__ x,
                                                float* __restrict__ mu,
                                                float* __restrict__ rsig) {
    int g = blockIdx.x;
    const float* xp = x + (size_t)g * 8 * S_;
    float s = 0.f, ss = 0.f;
    for (int i = threadIdx.x; i < 8 * S_; i += 256) {
        float v = xp[i];
        s += v; ss += v * v;
    }
    __shared__ float sh0[256];
    __shared__ float sh1[256];
    sh0[threadIdx.x] = s; sh1[threadIdx.x] = ss;
    __syncthreads();
    for (int off = 128; off > 0; off >>= 1) {
        if (threadIdx.x < off) {
            sh0[threadIdx.x] += sh0[threadIdx.x + off];
            sh1[threadIdx.x] += sh1[threadIdx.x + off];
        }
        __syncthreads();
    }
    if (threadIdx.x == 0) {
        float n = 8.f * (float)S_;
        float m = sh0[0] / n;
        float var = sh1[0] / n - m * m;
        mu[g] = m;
        rsig[g] = rsqrtf(var + EPSF);
    }
}

// ---------------------------------------------------------------------------
// Kernel 2: normalize + QKV projection (192x64 @ 64xS). Writes f16:
//   q_ws : [H][S][8]  (pre-scaled by D^-0.5)  -> WMMA A rows
//   k_ws : [H][S][8]                          -> WMMA B columns (K=d padded)
//   vT_ws: [H][8][S]  (d-major)               -> WMMA B rows packed in t
// ---------------------------------------------------------------------------
__global__ __launch_bounds__(256) void qkv_proj(const float* __restrict__ x,
                                                const float* __restrict__ gw,
                                                const float* __restrict__ gb,
                                                const float* __restrict__ qkv_w,
                                                const float* __restrict__ mu,
                                                const float* __restrict__ rsig,
                                                _Float16* __restrict__ qws,
                                                _Float16* __restrict__ kws,
                                                _Float16* __restrict__ vtws) {
    __shared__ float xn[64][128];
    int s0 = blockIdx.x * 128;
    for (int i = threadIdx.x; i < 64 * 128; i += 256) {
        int c = i >> 7, j = i & 127, g = c >> 3;
        xn[c][j] = (x[(size_t)c * S_ + s0 + j] - mu[g]) * rsig[g] * gw[c] + gb[c];
    }
    __syncthreads();
    const float scale = 0.35355339059327373f;  // 1/sqrt(D)
    for (int i = threadIdx.x; i < 192 * 128; i += 256) {
        int o = i >> 7, j = i & 127;
        const float* wr = qkv_w + o * 64;
        __builtin_prefetch(wr, 0, 1);  // global_prefetch_b8
        float acc = 0.f;
        #pragma unroll 8
        for (int c = 0; c < 64; ++c) acc += wr[c] * xn[c][j];
        int s = s0 + j;
        if (o < 64) {
            int h = o >> 3, d = o & 7;
            qws[((size_t)h * S_ + s) * 8 + d] = (_Float16)(acc * scale);
        } else if (o < 128) {
            int oo = o - 64; int h = oo >> 3, d = oo & 7;
            kws[((size_t)h * S_ + s) * 8 + d] = (_Float16)acc;
        } else {
            int oo = o - 128; int h = oo >> 3, d = oo & 7;
            vtws[((size_t)(h * 8 + d)) * S_ + s] = (_Float16)acc;
        }
    }
}

// ---------------------------------------------------------------------------
// Stage one 64-key chunk (K: 1KB [t][d], V: 1KB [d][t]) into LDS.
// wv is wave-uniform (readfirstlane) so the branch is scalar and each wave
// issues exactly one global_load_async_to_lds_b64: waves 0-3 fill K, 4-7 V.
// ---------------------------------------------------------------------------
__device__ __forceinline__ void stage_chunk(const _Float16* __restrict__ kws,
                                            const _Float16* __restrict__ vtws,
                                            _Float16* kdst, _Float16* vdst,
                                            int h, int kb, int wv, int lane) {
#if HAS_ASYNC_LDS
    if (wv < 4) {
        int i = wv * 32 + lane;                         // 0..127, 8B each
        __builtin_amdgcn_global_load_async_to_lds_b64(
            GPTR64(kws + ((size_t)h * S_ + kb) * 8 + i * 4),
            LPTR64(kdst + i * 4), 0, 0);
    } else {
        int j = (wv - 4) * 32 + lane;                   // 0..127
        int d = j >> 4, c4 = (j & 15) * 4;
        __builtin_amdgcn_global_load_async_to_lds_b64(
            GPTR64(vtws + (size_t)(h * 8 + d) * S_ + kb + c4),
            LPTR64(vdst + d * 64 + c4), 0, 0);
    }
#else
    if (wv < 4) {
        int i = wv * 32 + lane;
        *(uint2*)(kdst + i * 4) =
            *(const uint2*)(kws + ((size_t)h * S_ + kb) * 8 + i * 4);
    } else {
        int j = (wv - 4) * 32 + lane;
        int d = j >> 4, c4 = (j & 15) * 4;
        *(uint2*)(vdst + d * 64 + c4) =
            *(const uint2*)(vtws + (size_t)(h * 8 + d) * S_ + kb + c4);
    }
#endif
}

// ---------------------------------------------------------------------------
// Kernel 3: flash attention. Grid = H * (S/128) blocks, 256 threads = 8 waves.
// Each wave owns a 16-query tile; loops over 64-key chunks with a
// triple-buffered ASYNCcnt-tracked LDS pipeline (prefetch distance 1;
// 3 buffers make the single-barrier-per-iteration schedule race-free:
// the buffer overwritten by stage(it+1) was last read at compute(it-2),
// which barrier(it-1) proves complete for every wave).
// Scores: v_wmma_f32_16x16x32_f16 with K padded 8->32. P converted
// C-layout -> A-layout through per-wave LDS staging (DS in-order per wave).
// ---------------------------------------------------------------------------
__global__ __launch_bounds__(256) void flash_attn(const _Float16* __restrict__ qws,
                                                  const _Float16* __restrict__ kws,
                                                  const _Float16* __restrict__ vtws,
                                                  float* __restrict__ attn) {
    __shared__ __align__(16) _Float16 kc[3][64 * 8];    // [buf][t][d]   3KB
    __shared__ __align__(16) _Float16 vc[3][8 * 64];    // [buf][d][t]   3KB
    __shared__ __align__(16) _Float16 pl[8 * 16 * 64];  // [wave][m][t] 16KB

    int tid  = threadIdx.x;
    int wave = tid >> 5;
    int lane = tid & 31;
    int wv   = __builtin_amdgcn_readfirstlane(wave);  // scalar wave id
    int lh   = lane >> 4;     // lane half (0: lanes0-15, 1: lanes16-31)
    int ln   = lane & 15;
    int h    = blockIdx.x >> 5;
    int qb   = blockIdx.x & 31;
    int q0   = qb * 128 + wave * 16;

    // ---- load Q A-tile: lanes 0-15 carry K=0..15 (d 0..7 real), rest zero
    v16h qa;
    #pragma unroll
    for (int e = 0; e < 16; ++e) qa[e] = (_Float16)0.f;
    if (lh == 0) {
        v8h qv = *(const v8h*)(qws + ((size_t)h * S_ + q0 + ln) * 8);
        #pragma unroll
        for (int e = 0; e < 8; ++e) qa[e] = qv[e];
    }

    v8f acc;
    #pragma unroll
    for (int r = 0; r < 8; ++r) acc[r] = 0.f;
    float M[8], L[8];
    #pragma unroll
    for (int r = 0; r < 8; ++r) { M[r] = -1e30f; L[r] = 0.f; }

    _Float16* pw = pl + wave * 16 * 64;
    const int NIT = S_ / 64;

    // prologue: stage chunk 0 into buffer 0
    stage_chunk(kws, vtws, kc[0], vc[0], h, 0, wv, lane);

    int bcur = 0;
    for (int it = 0; it < NIT; ++it) {
        int bnext = bcur + 1;
        if (bnext == 3) bnext = 0;

        // prefetch next chunk into the third buffer, then wait for current
        if (it + 1 < NIT) {
            stage_chunk(kws, vtws, kc[bnext], vc[bnext], h, (it + 1) * 64, wv, lane);
            WAIT_ASYNC(1);
        } else {
            WAIT_ASYNC(0);
        }
        __syncthreads();

        const _Float16* kcur = kc[bcur];
        const _Float16* vcur = vc[bcur];

        // ---- 4 score tiles (16 queries x 64 keys), K=32 padded from d=8
        v8f sc[4];
        #pragma unroll
        for (int kt = 0; kt < 4; ++kt) {
            v16h kbm;
            #pragma unroll
            for (int e = 0; e < 16; ++e) kbm[e] = (_Float16)0.f;
            if (lh == 0) {
                v8h kv = *(const v8h*)(kcur + (kt * 16 + ln) * 8);
                #pragma unroll
                for (int e = 0; e < 8; ++e) kbm[e] = kv[e];
            }
            v8f z;
            #pragma unroll
            for (int r = 0; r < 8; ++r) z[r] = 0.f;
            sc[kt] = __builtin_amdgcn_wmma_f32_16x16x32_f16(
                false, qa, false, kbm, (short)0, z, false, false);
        }

        // ---- online softmax (row = r + 8*lh, columns striped over 16 lanes)
        float nm[8], rs[8];
        #pragma unroll
        for (int r = 0; r < 8; ++r) {
            float mx = fmaxf(fmaxf(sc[0][r], sc[1][r]), fmaxf(sc[2][r], sc[3][r]));
            for (int m = 1; m < 16; m <<= 1) mx = fmaxf(mx, __shfl_xor(mx, m, 32));
            nm[r] = fmaxf(M[r], mx);
        }
        #pragma unroll
        for (int r = 0; r < 8; ++r) {
            float sum = 0.f;
            #pragma unroll
            for (int kt = 0; kt < 4; ++kt) {
                float p = __expf(sc[kt][r] - nm[r]);
                sc[kt][r] = p;
                sum += p;
            }
            for (int m = 1; m < 16; m <<= 1) sum += __shfl_xor(sum, m, 32);
            rs[r] = sum;
        }
        #pragma unroll
        for (int r = 0; r < 8; ++r) {
            float corr = __expf(M[r] - nm[r]);
            acc[r] *= corr;
            L[r] = L[r] * corr + rs[r];
            M[r] = nm[r];
        }

        // ---- P: C-layout -> [m][t] f16 in per-wave LDS (wave-synchronous)
        #pragma unroll
        for (int kt = 0; kt < 4; ++kt)
            #pragma unroll
            for (int r = 0; r < 8; ++r)
                pw[(r + 8 * lh) * 64 + kt * 16 + ln] = (_Float16)sc[kt][r];
        asm volatile("" ::: "memory");  // DS is in-order per wave; block reordering

        // ---- O += P(16x64) x V(64x16): two K=32 WMMAs
        #pragma unroll
        for (int ka = 0; ka < 2; ++ka) {
            v16h pa;
            #pragma unroll
            for (int i = 0; i < 8; ++i) {
                int t0 = 32 * ka + (i < 4 ? lh * 8 + 2 * i : 16 + lh * 8 + 2 * (i - 4));
                h2 pr = *(const h2*)(pw + ln * 64 + t0);
                pa[2 * i] = pr[0]; pa[2 * i + 1] = pr[1];
            }
            v16h vb;
            if (ln < 8) {
                #pragma unroll
                for (int i = 0; i < 8; ++i) {
                    int tl = 32 * ka + (i < 4 ? lh * 8 + 2 * i : 16 + lh * 8 + 2 * (i - 4));
                    h2 vv = *(const h2*)(vcur + ln * 64 + tl);
                    vb[2 * i] = vv[0]; vb[2 * i + 1] = vv[1];
                }
            } else {
                #pragma unroll
                for (int e = 0; e < 16; ++e) vb[e] = (_Float16)0.f;
            }
            acc = __builtin_amdgcn_wmma_f32_16x16x32_f16(
                false, pa, false, vb, (short)0, acc, false, false);
        }

        bcur = bnext;
    }

    // ---- write normalized O: C-layout (m = r+8*lh, n = d = ln)
    if (ln < 8) {
        #pragma unroll
        for (int r = 0; r < 8; ++r) {
            int m = r + 8 * lh;
            attn[(size_t)(h * 8 + ln) * S_ + q0 + m] = acc[r] / L[r];
        }
    }
}

// ---------------------------------------------------------------------------
// Kernel 4: out projection (64x64) + bias + residual.
// ---------------------------------------------------------------------------
__global__ __launch_bounds__(256) void out_proj(const float* __restrict__ attn,
                                                const float* __restrict__ ow,
                                                const float* __restrict__ ob,
                                                const float* __restrict__ x,
                                                float* __restrict__ out) {
    int idx = blockIdx.x * 256 + threadIdx.x;
    int o = idx >> 12, s = idx & 4095;
    const float* wr = ow + o * 64;
    __builtin_prefetch(wr, 0, 1);
    float acc = ob[o];
    #pragma unroll 8
    for (int c = 0; c < 64; ++c) acc += wr[c] * attn[(size_t)c * S_ + s];
    out[idx] = acc + x[idx];
}

// ---------------------------------------------------------------------------
extern "C" void kernel_launch(void* const* d_in, const int* in_sizes, int n_in,
                              void* d_out, int out_size, void* d_ws, size_t ws_size,
                              hipStream_t stream) {
    const float* x     = (const float*)d_in[0];
    const float* gw    = (const float*)d_in[1];
    const float* gb    = (const float*)d_in[2];
    const float* qkv_w = (const float*)d_in[3];
    const float* ow    = (const float*)d_in[4];
    const float* ob    = (const float*)d_in[5];
    float* out = (float*)d_out;

    // workspace layout
    char* ws = (char*)d_ws;
    float*    mu   = (float*)ws;                         // 8 f32
    float*    rsig = mu + 8;                             // 8 f32
    _Float16* qws  = (_Float16*)(ws + 256);              // H*S*8 f16 = 512KB
    _Float16* kws  = qws + (size_t)H_ * S_ * 8;          // 512KB
    _Float16* vtws = kws + (size_t)H_ * S_ * 8;          // 512KB
    float*    attn = (float*)(vtws + (size_t)H_ * S_ * 8);  // C*S f32 = 1MB

    gn_stats  <<<G_,            256, 0, stream>>>(x, mu, rsig);
    qkv_proj  <<<S_ / 128,      256, 0, stream>>>(x, gw, gb, qkv_w, mu, rsig,
                                                  qws, kws, vtws);
    flash_attn<<<H_ * (S_/128), 256, 0, stream>>>(qws, kws, vtws, attn);
    out_proj  <<<(C_ * S_)/256, 256, 0, stream>>>(attn, ow, ob, x, out);
}
